// RecurrentConvBlock_52372831207963
// MI455X (gfx1250) — matrix operand order repaired and checked by reading, not compiled, not run on hardware
//
#include <hip/hip_runtime.h>

// ---------------------------------------------------------------------------
// CDNA5 (gfx1250) ConvLSTM2D stack: fully fused per-timestep step kernel.
// Implicit-GEMM bf16 WMMA (v_wmma_f32_16x16x32_bf16), f32 accumulation,
// LSTM gate math fused in-kernel via a 128x384 LDS exchange buffer.
// ---------------------------------------------------------------------------

typedef __attribute__((ext_vector_type(16))) __bf16 v16bf;
typedef __attribute__((ext_vector_type(8)))  __bf16 bf16x8;
typedef __attribute__((ext_vector_type(8)))  float  v8f;

constexpr int IH   = 96;        // spatial H
constexpr int IW   = 96;        // spatial W
constexpr int IHW  = IH * IW;   // 9216
constexpr int FCH  = 96;        // hidden features F
constexpr int NCH  = 384;       // 4*F gate channels
constexpr int MTILE = 128;      // pixel tile
constexpr int LDT  = 40;        // padded LDS row stride (bf16 elems, 80B)
constexpr int NB   = 8;         // batch
constexpr int NT   = 10;        // timesteps

static __device__ __forceinline__ v16bf combine8(bf16x8 lo, bf16x8 hi) {
  v16bf r;
#pragma unroll
  for (int e = 0; e < 8; ++e) { r[e] = lo[e]; r[8 + e] = hi[e]; }
  return r;
}

// One convolution phase: acc[2][6] += implicit-GEMM of img (*) wt.
// img: bf16 NHWC with per-batch stride `bstride` (elements)
// wt : bf16 pre-transposed/padded [KH*KH][NCH][CPAD]
template <int KH, int CIN, int CPAD>
static __device__ __forceinline__ void conv_phase(
    const __bf16* __restrict__ img, long bstride,
    const __bf16* __restrict__ wt,
    __bf16* As, __bf16* Bs, v8f (&acc)[2][6],
    int m0, int tid, int wm, int wn, int half, int l15) {
  const int srow = tid >> 2;                 // 0..127 (pixel row / weight row)
  const int sseg = (tid & 3) * 8;            // 0,8,16,24 channel sub-seg
  const int p  = m0 + srow;
  const int b  = p / IHW;
  const int q  = p - b * IHW;
  const int py = q / IW;
  const int px = q - py * IW;
  constexpr int PAD = KH / 2;
  constexpr int KCH = CPAD / 32;

#pragma unroll 1
  for (int tap = 0; tap < KH * KH; ++tap) {
    const int dy = tap / KH - PAD;
    const int dx = tap % KH - PAD;
    const int y = py + dy, x = px + dx;
    const bool inb = (y >= 0) & (y < IH) & (x >= 0) & (x < IW);
    const __bf16* srcA = img + (long)b * bstride + ((long)y * IW + x) * CIN;

#pragma unroll
    for (int kc = 0; kc < KCH; ++kc) {
      __syncthreads();                       // LDS reuse fence
      // ---- stage A: 128 pixels x 32 channels (zero-pad OOB / chan-pad) ----
      {
        const int c0 = kc * 32 + sseg;
        __bf16* dst = &As[srow * LDT + sseg];
        if (inb && (c0 + 8) <= CIN) {
          *(bf16x8*)dst = *(const bf16x8*)(srcA + c0);
        } else {
#pragma unroll
          for (int e = 0; e < 8; ++e)
            dst[e] = (inb && (c0 + e) < CIN) ? srcA[c0 + e] : (__bf16)0.f;
        }
      }
      // ---- stage B: 384 couts x 32 channels (pre-padded) ----
#pragma unroll
      for (int r = 0; r < 3; ++r) {
        const int n = srow + r * 128;
        const __bf16* srcB = wt + ((long)tap * NCH + n) * CPAD + kc * 32 + sseg;
        *(bf16x8*)&Bs[n * LDT + sseg] = *(const bf16x8*)srcB;
      }
      __syncthreads();

      // ---- one K=32 step: 2 A-frags x 6 B-frags -> 12 WMMAs / wave ----
      // A 16x32 bf16: lane<16: M=l, K 0..7 & 16..23; lane>=16: K 8..15 & 24..31
      v16bf afrag[2];
#pragma unroll
      for (int i = 0; i < 2; ++i) {
        const __bf16* ap = &As[(wm * 32 + i * 16 + l15) * LDT + half * 8];
        afrag[i] = combine8(*(const bf16x8*)ap, *(const bf16x8*)(ap + 16));
      }
      // B 32x16 bf16: lane<16: N=l, K 0..7 & 16..23; lane>=16: K 8..15 & 24..31
#pragma unroll
      for (int j = 0; j < 6; ++j) {
        const __bf16* bp = &Bs[(wn * 96 + j * 16 + l15) * LDT + half * 8];
        const v16bf bfrag = combine8(*(const bf16x8*)bp, *(const bf16x8*)(bp + 16));
#pragma unroll
        for (int i = 0; i < 2; ++i)
          acc[i][j] = __builtin_amdgcn_wmma_f32_16x16x32_bf16(
              false, afrag[i], false, bfrag, (short)0, acc[i][j], false, false);
      }
    }
  }
}

// Fully fused ConvLSTM timestep: g = conv(x_t,wx)+conv(h,wh)+b; LSTM update.
// Block: 512 threads = 16 waves (4 M-waves x 4 N-waves), tile 128 x 384.
template <int KH, int CX, int CXP>
__global__ __launch_bounds__(512) void convlstm_step_kernel(
    const __bf16* __restrict__ xin, long xbstride,
    const __bf16* __restrict__ wxt,
    const __bf16* __restrict__ hin,
    const __bf16* __restrict__ wht,
    const float* __restrict__ bias,
    float* __restrict__ Cst, __bf16* __restrict__ Hst,
    __bf16* __restrict__ outbf,
    float* __restrict__ outf32, long obstride,
    int t0) {
  // 192KB LDS: A/B staging tiles aliased with the 128x384 f32 gate exchange.
  __shared__ alignas(128) char smem[MTILE * NCH * 4];
  __bf16* As = (__bf16*)smem;
  __bf16* Bs = (__bf16*)(smem + MTILE * LDT * 2);
  float*  Xg = (float*)smem;

  const int m0   = blockIdx.x * MTILE;
  const int tid  = threadIdx.x;
  const int lane = tid & 31;
  const int wave = tid >> 5;                 // 0..15
  const int wm   = wave >> 2;                // 0..3 -> 32 pixel rows each
  const int wn   = wave & 3;                 // 0..3 -> 96 gate channels each
  const int half = lane >> 4;
  const int l15  = lane & 15;

  v8f acc[2][6];
#pragma unroll
  for (int i = 0; i < 2; ++i)
#pragma unroll
    for (int j = 0; j < 6; ++j) { v8f z = {}; acc[i][j] = z; }

  // x-conv phase (input-to-gates)
  conv_phase<KH, CX, CXP>(xin, xbstride, wxt, As, Bs, acc,
                          m0, tid, wm, wn, half, l15);
  // h-conv phase (recurrent); h == 0 at t0, so skip entirely
  if (!t0)
    conv_phase<KH, FCH, FCH>(hin, (long)IHW * FCH, wht, As, Bs, acc,
                             m0, tid, wm, wn, half, l15);

  __syncthreads();   // all waves done reading As/Bs before aliasing as Xg

  // ---- scatter gate preactivations (+bias) to LDS exchange ----
  // C/D layout: VGPR r, lane<16 -> M=r, lane>=16 -> M=8+r; N = lane&15.
#pragma unroll
  for (int i = 0; i < 2; ++i) {
    const int mrow = wm * 32 + i * 16 + half * 8;
#pragma unroll
    for (int j = 0; j < 6; ++j) {
      const int n = wn * 96 + j * 16 + l15;
      const float bn = bias[n];
#pragma unroll
      for (int r = 0; r < 8; ++r)
        Xg[(mrow + r) * NCH + n] = acc[i][j][r] + bn;
    }
  }
  __syncthreads();

  // ---- fused LSTM pointwise update: 128 pixels x 96 features ----
#pragma unroll
  for (int k = 0; k < (MTILE * FCH) / 512; ++k) {  // 24 iters
    const int ii = k * 512 + tid;
    const int m = ii / FCH;
    const int f = ii - m * FCH;
    const float gi = Xg[m * NCH + f];
    const float gf = Xg[m * NCH + FCH + f];
    const float gc = Xg[m * NCH + 2 * FCH + f];
    const float go = Xg[m * NCH + 3 * FCH + f];
    const long p = m0 + m;
    const long sidx = p * (long)FCH + f;
    const float cprev = t0 ? 0.f : Cst[sidx];
    auto hsg = [](float v) { return fminf(fmaxf(0.2f * v + 0.5f, 0.f), 1.f); };
    const float c = hsg(gf) * cprev + hsg(gi) * tanhf(gc);
    Cst[sidx] = c;
    const float h = hsg(go) * tanhf(c);
    Hst[sidx] = (__bf16)h;
    const float hr = fmaxf(h, 0.f);
    if (outbf) outbf[sidx] = (__bf16)hr;
    if (outf32) {
      const long bb = p / IHW;
      const long qq = p - bb * IHW;
      outf32[bb * obstride + qq * FCH + f] = hr;
    }
  }
}

__global__ void cvt_f32_bf16_kernel(const float* __restrict__ in,
                                    __bf16* __restrict__ out, long n) {
  long i = (long)blockIdx.x * blockDim.x + threadIdx.x;
  if (i < n) out[i] = (__bf16)in[i];
}

// Repack conv weight [KH,KW,Cin,384] f32 -> [taps][384][Cpad] bf16 (zero-pad).
__global__ void prep_weight_kernel(const float* __restrict__ w,
                                   __bf16* __restrict__ wt,
                                   int taps, int Cin, int Cpad) {
  long i = (long)blockIdx.x * blockDim.x + threadIdx.x;
  const long tot = (long)taps * NCH * Cpad;
  if (i >= tot) return;
  const int c = (int)(i % Cpad);
  const long r = i / Cpad;
  const int n = (int)(r % NCH);
  const int tap = (int)(r / NCH);
  wt[i] = (c < Cin) ? (__bf16)w[((long)tap * Cin + c) * NCH + n] : (__bf16)0.f;
}

extern "C" void kernel_launch(void* const* d_in, const int* in_sizes, int n_in,
                              void* d_out, int out_size, void* d_ws, size_t ws_size,
                              hipStream_t stream) {
  (void)in_sizes; (void)n_in; (void)out_size; (void)ws_size;
  const float* X   = (const float*)d_in[0];
  const float* wx1 = (const float*)d_in[1];
  const float* wh1 = (const float*)d_in[2];
  const float* b1  = (const float*)d_in[3];
  const float* wx2 = (const float*)d_in[4];
  const float* wh2 = (const float*)d_in[5];
  const float* b2  = (const float*)d_in[6];
  float* out = (float*)d_out;

  const int Cin = 48;
  const long BHW = (long)NB * IHW;          // 73728, divisible by MTILE

  // ---- workspace carve (256B aligned) ----
  char* wsp = (char*)d_ws;
  auto carve = [&](size_t bytes) {
    char* pp = wsp;
    wsp += (bytes + 255) & ~(size_t)255;
    return pp;
  };
  __bf16* Xbf  = (__bf16*)carve((size_t)NB * NT * IHW * Cin * 2);  // ~71 MB
  __bf16* Wx1t = (__bf16*)carve((size_t)25 * NCH * 64 * 2);
  __bf16* Wh1t = (__bf16*)carve((size_t)25 * NCH * 96 * 2);
  __bf16* Wx2t = (__bf16*)carve((size_t)9 * NCH * 96 * 2);
  __bf16* Wh2t = (__bf16*)carve((size_t)9 * NCH * 96 * 2);
  float*  Cst  = (float*)carve((size_t)BHW * FCH * 4);             // ~28 MB
  __bf16* Hst  = (__bf16*)carve((size_t)BHW * FCH * 2);            // ~14 MB
  __bf16* H1   = (__bf16*)carve((size_t)NT * BHW * FCH * 2);       // ~142 MB

  // ---- prep: bf16 conversion + weight repack ----
  {
    const long nX = (long)NB * NT * IHW * Cin;
    cvt_f32_bf16_kernel<<<(int)((nX + 255) / 256), 256, 0, stream>>>(X, Xbf, nX);
    long n;
    n = 25L * NCH * 64;
    prep_weight_kernel<<<(int)((n + 255) / 256), 256, 0, stream>>>(wx1, Wx1t, 25, 48, 64);
    n = 25L * NCH * 96;
    prep_weight_kernel<<<(int)((n + 255) / 256), 256, 0, stream>>>(wh1, Wh1t, 25, 96, 96);
    n = 9L * NCH * 96;
    prep_weight_kernel<<<(int)((n + 255) / 256), 256, 0, stream>>>(wx2, Wx2t, 9, 96, 96);
    prep_weight_kernel<<<(int)((n + 255) / 256), 256, 0, stream>>>(wh2, Wh2t, 9, 96, 96);
  }

  const int grid = (int)(BHW / MTILE);  // 576 blocks x 512 threads

  // ---- layer 1: 5x5 ConvLSTM (48 -> 96), ReLU(h) -> H1 sequence (bf16) ----
  for (int t = 0; t < NT; ++t) {
    convlstm_step_kernel<5, 48, 64><<<grid, 512, 0, stream>>>(
        Xbf + (long)t * IHW * Cin, (long)NT * IHW * Cin, Wx1t,
        Hst, Wh1t, b1, Cst, Hst,
        H1 + (long)t * BHW * FCH, nullptr, 0, t == 0 ? 1 : 0);
  }
  // ---- layer 2: 3x3 ConvLSTM (96 -> 96), ReLU(h) -> d_out [B,T,H,W,F] ----
  for (int t = 0; t < NT; ++t) {
    convlstm_step_kernel<3, 96, 96><<<grid, 512, 0, stream>>>(
        H1 + (long)t * BHW * FCH, (long)IHW * FCH, Wx2t,
        Hst, Wh2t, b2, Cst, Hst,
        nullptr, out + (long)t * IHW * FCH, (long)NT * IHW * FCH,
        t == 0 ? 1 : 0);
  }
}
